// NystromAttention_77970836291948
// MI455X (gfx1250) — matrix operand order repaired and checked by reading, not compiled, not run on hardware
//
#include <hip/hip_runtime.h>
#include <hip/hip_bf16.h>

// ---------------------------------------------------------------------------
// Nystrom attention for MI455X (gfx1250).  Memory-bound (~3 GB traffic vs
// ~10.7 GFLOP -> ~130 us floor at 23.3 TB/s): one pass over Q/K for landmarks,
// one K/V pass (split-S flash) for c = softmax(QL K^T) V, one Q pass for the
// output.  All big matmuls: v_wmma_f32_16x16x32_f16 (f16 operands, f32 acc).
// All WMMA operands are contiguous 16B LDS fragments (ds_load_b128, no cvt in
// the operand path); all global staging uses float4 (b128) accesses.
// ---------------------------------------------------------------------------

typedef _Float16 v16h __attribute__((ext_vector_type(16)));
typedef float    v8f  __attribute__((ext_vector_type(8)));

constexpr int   kS  = 4096;
constexpr int   kD  = 64;
constexpr int   kL  = 64;
constexpr int   kH  = 8;
constexpr int   kBH = 64;          // B*H
constexpr int   kNS = 8;           // S-splits for kernel_3 @ V
constexpr int   kSP = kS / kNS;    // 512 rows per split
constexpr float kScale = 0.35355339059327373f;   // 64^(-1/4)

constexpr int kLDH = 72;           // f16 tile stride (144 B rows: 16B aligned, conflict-free)
constexpr int kLDL = 536;          // f16 logits stride (1072 B rows: 16B aligned, distinct banks)
constexpr int kLDF = 68;           // f32 tile stride (multiple of 4 -> float4-clean)

// workspace offsets (in floats)
constexpr size_t WS_QL  = 0;
constexpr size_t WS_KL  = WS_QL + (size_t)kBH * kL * kD;
constexpr size_t WS_A2  = WS_KL + (size_t)kBH * kL * kD;
constexpr size_t WS_INV = WS_A2 + (size_t)kBH * kL * kL;
constexpr size_t WS_C   = WS_INV + (size_t)kBH * kL * kL;
constexpr size_t WS_CP  = WS_C  + (size_t)kBH * kL * kD;
constexpr size_t WS_MS  = WS_CP + (size_t)kBH * kNS * kL * kD;
constexpr size_t WS_LS  = WS_MS + (size_t)kBH * kNS * kL;
constexpr size_t WS_GM  = WS_LS + (size_t)kBH * kNS * kL;   // 1 float scalar

#define WMMA_F16(A, B, C) \
  __builtin_amdgcn_wmma_f32_16x16x32_f16(false, (A), false, (B), (short)0, (C), false, false)

// ---- WMMA operand loaders (A and B share one contiguous k order) -------------------
// A (16x32 f16): lane m=l&15, hi=l>>4; elem j -> K = 16*hi + j
__device__ __forceinline__ v16h ldA_f16(const _Float16* p, int ld, int lane) {
  const int m = lane & 15, hb = (lane >> 4) * 16;
  const _Float16* r = p + m * ld;
  v16h a;
#pragma unroll
  for (int j = 0; j < 16; ++j) a[j] = r[hb + j];
  return a;
}
// B (32x16 f16): lane n=l&15; elem j -> K = 16*hi + j.  Source must hold
// B^T row-major (i.e. p[n*ld + k] == B[k][n]) -> contiguous 16B loads.
__device__ __forceinline__ v16h ldBT_f16(const _Float16* p, int ld, int lane) {
  const int n = lane & 15, kb = (lane >> 4) * 16;
  const _Float16* r = p + n * ld + kb;
  v16h b;
#pragma unroll
  for (int j = 0; j < 16; ++j) b[j] = r[j];
  return b;
}
// C/D f32: vgpr r -> row r + 8*hi, col = lane&15
__device__ __forceinline__ void stC_f32(float* p, int ld, int lane, v8f acc) {
  const int n = lane & 15, mo = (lane >> 4) * 8;
#pragma unroll
  for (int r = 0; r < 8; ++r) p[(r + mo) * ld + n] = acc[r];
}
__device__ __forceinline__ void stC_f16(_Float16* p, int ld, int lane, v8f acc) {
  const int n = lane & 15, mo = (lane >> 4) * 8;
#pragma unroll
  for (int r = 0; r < 8; ++r) p[(r + mo) * ld + n] = (_Float16)acc[r];
}

// ---------------------------------------------------------------------------
// Kernel 1: landmark pooling.  grid = BH*L blocks, 64 threads (d = tid).
// ---------------------------------------------------------------------------
__global__ __launch_bounds__(64) void k_landmarks(const float* __restrict__ Q,
                                                  const float* __restrict__ K,
                                                  const float* __restrict__ mask,
                                                  float* __restrict__ QL,
                                                  float* __restrict__ KL) {
  const int bh = blockIdx.x >> 6;
  const int l  = blockIdx.x & 63;
  const int b  = bh / kH;
  const int d  = threadIdx.x;
  const size_t base = ((size_t)bh * kS + (size_t)l * 64) * kD + d;
  const float* mk = mask + (size_t)b * kS + l * 64;
  float sq = 0.f, sk = 0.f;
#pragma unroll 4
  for (int r = 0; r < 64; ++r) {
    const float m = mk[r];
    sq += Q[base + (size_t)r * kD] * m;
    sk += K[base + (size_t)r * kD] * m;
  }
  const float c = kScale * (1.0f / 64.0f);
  QL[(size_t)bh * kL * kD + l * kD + d] = sq * c;
  KL[(size_t)bh * kL * kD + l * kD + d] = sk * c;
}

// ---------------------------------------------------------------------------
// Kernel 2a: A2 = softmax_rows(QL @ KL^T); global max of column sums -> gm.
// ---------------------------------------------------------------------------
__global__ __launch_bounds__(256) void k_kernel2(const float* __restrict__ QL,
                                                 const float* __restrict__ KL,
                                                 float* __restrict__ A2,
                                                 int* __restrict__ gm) {
  extern __shared__ char smem[];
  float* sQ = (float*)smem;           // 64x66
  float* sK = sQ + 64 * 66;
  float* sA = sK + 64 * 66;
  float* sc = sA + 64 * 66;           // 64 colsums
  const int bh = blockIdx.x, t = threadIdx.x;
#pragma unroll
  for (int n = 0; n < 16; ++n) {
    const int idx = t + 256 * n, i = idx >> 6, j = idx & 63;
    sQ[i * 66 + j] = QL[(size_t)bh * 4096 + idx];
    sK[i * 66 + j] = KL[(size_t)bh * 4096 + idx];
  }
  __syncthreads();
  {
    const int j = t & 63, i0 = (t >> 6) * 16;
    for (int i = i0; i < i0 + 16; ++i) {
      float s = 0.f;
#pragma unroll 8
      for (int k = 0; k < 64; ++k) s = fmaf(sQ[i * 66 + k], sK[j * 66 + k], s);
      sA[i * 66 + j] = s;
    }
  }
  __syncthreads();
  if (t < 64) {                        // row softmax
    float mx = -1e30f;
    for (int j = 0; j < 64; ++j) mx = fmaxf(mx, sA[t * 66 + j]);
    float sm = 0.f;
    for (int j = 0; j < 64; ++j) { float e = __expf(sA[t * 66 + j] - mx); sA[t * 66 + j] = e; sm += e; }
    const float inv = 1.0f / sm;
    for (int j = 0; j < 64; ++j) sA[t * 66 + j] *= inv;
  }
  __syncthreads();
  if (t < 64) {                        // column sums
    float cs = 0.f;
    for (int i = 0; i < 64; ++i) cs += sA[i * 66 + t];
    sc[t] = cs;
  }
  __syncthreads();
  if (t == 0) {
    float mx = sc[0];
    for (int j = 1; j < 64; ++j) mx = fmaxf(mx, sc[j]);
    atomicMax(gm, __float_as_int(mx));   // colsums > 0 -> int order == float order
  }
#pragma unroll
  for (int n = 0; n < 16; ++n) {
    const int idx = t + 256 * n, i = idx >> 6, j = idx & 63;
    A2[(size_t)bh * 4096 + idx] = sA[i * 66 + j];
  }
}

// ---------------------------------------------------------------------------
// Kernel 2b: Newton-Schulz pseudo-inverse, 6 iterations, exact f32.
// ---------------------------------------------------------------------------
__device__ __forceinline__ void mm64(const float* A, const float* B, float* C, int t) {
  const int j = t & 63, i0 = (t >> 6) * 16;
  for (int i = i0; i < i0 + 16; ++i) {
    float s = 0.f;
#pragma unroll 8
    for (int k = 0; k < 64; ++k) s = fmaf(A[i * 66 + k], B[k * 66 + j], s);
    C[i * 66 + j] = s;
  }
}
__device__ __forceinline__ void ew_isub(float coef, const float* M, float* O, int t) {
#pragma unroll
  for (int n = 0; n < 16; ++n) {
    const int idx = t + 256 * n, i = idx >> 6, j = idx & 63;
    O[i * 66 + j] = (i == j ? coef : 0.f) - M[i * 66 + j];
  }
}
__global__ __launch_bounds__(256) void k_inverse(const float* __restrict__ A2,
                                                 const float* __restrict__ gm,
                                                 float* __restrict__ INV) {
  extern __shared__ char smem[];
  float* Km = (float*)smem;
  float* Vm = Km + 64 * 66;
  float* M1 = Vm + 64 * 66;
  float* M2 = M1 + 64 * 66;
  float* M3 = M2 + 64 * 66;
  const int bh = blockIdx.x, t = threadIdx.x;
#pragma unroll
  for (int n = 0; n < 16; ++n) {
    const int idx = t + 256 * n, i = idx >> 6, j = idx & 63;
    Km[i * 66 + j] = A2[(size_t)bh * 4096 + idx];
  }
  __syncthreads();
  const float invg = 1.0f / gm[0];
#pragma unroll
  for (int n = 0; n < 16; ++n) {
    const int idx = t + 256 * n, i = idx >> 6, j = idx & 63;
    Vm[i * 66 + j] = Km[j * 66 + i] * invg;           // Km^T / max(colsum)
  }
  __syncthreads();
  for (int it = 0; it < 6; ++it) {
    mm64(Km, Vm, M1, t); __syncthreads();             // KV
    ew_isub(7.0f, M1, M2, t); __syncthreads();
    mm64(M1, M2, M3, t); __syncthreads();             // KV(7I-KV)
    ew_isub(15.0f, M3, M2, t); __syncthreads();
    mm64(M1, M2, M3, t); __syncthreads();             // KV(15I-...)
    ew_isub(13.0f, M3, M2, t); __syncthreads();
    mm64(Vm, M2, M3, t); __syncthreads();             // Vm(13I-...)
#pragma unroll
    for (int n = 0; n < 16; ++n) {
      const int idx = t + 256 * n, i = idx >> 6, j = idx & 63;
      Vm[i * 66 + j] = 0.25f * M3[i * 66 + j];
    }
    __syncthreads();
  }
#pragma unroll
  for (int n = 0; n < 16; ++n) {
    const int idx = t + 256 * n, i = idx >> 6, j = idx & 63;
    INV[(size_t)bh * 4096 + idx] = Vm[i * 66 + j];
  }
}

// ---------------------------------------------------------------------------
// Kernel 3: partial c = exp(QL@K^T - m) @ V over one S-split (flash style).
// grid (BH, NS), 256 thr (8 waves x two 16x16 tiles).  f16 logit fragments,
// transposed V staging -> all WMMA operands are contiguous b128 LDS loads.
// ---------------------------------------------------------------------------
__global__ __launch_bounds__(256) void k_kernel3(const float* __restrict__ K,
                                                 const float* __restrict__ V,
                                                 const float* __restrict__ mask,
                                                 const float* __restrict__ QL,
                                                 float* __restrict__ CP,
                                                 float* __restrict__ MS,
                                                 float* __restrict__ LS) {
  extern __shared__ char smem[];
  _Float16* sLg  = (_Float16*)smem;                         // 64 x kLDL (f16 logits/P)
  float*    sStg = (float*)smem;                            // reuse: 64 x kLDF f32 staging
  float*    sRed = (float*)(smem + 64 * kLDL * 2);          // 64 x 4
  float*    sM   = sRed + 256;                              // 64
  float*    sL   = sM + 64;                                 // 64
  _Float16* sQL  = (_Float16*)(sL + 64);                    // 64 x 72
  _Float16* sKV  = sQL + 64 * kLDH;                         // 64 x 72 (K row-major / V^T)
  const int bh = blockIdx.x, sp = blockIdx.y, b = bh / kH;
  const int t = threadIdx.x, lane = t & 31, wave = t >> 5;
  const int tn = wave & 3, tmB = (wave >> 2) * 2;

  const float4* Kb4 = (const float4*)(K + ((size_t)bh * kS + (size_t)sp * kSP) * kD);
  const float4* Vb4 = (const float4*)(V + ((size_t)bh * kS + (size_t)sp * kSP) * kD);
  const float4* QL4 = (const float4*)(QL + (size_t)bh * 4096);

#pragma unroll
  for (int n = 0; n < 4; ++n) {                             // stage QL (row-major f16)
    const int i4 = t + 256 * n, row = i4 >> 4, c4 = (i4 & 15) * 4;
    const float4 v = QL4[i4];
    _Float16* d = sQL + row * kLDH + c4;
    d[0] = (_Float16)v.x; d[1] = (_Float16)v.y; d[2] = (_Float16)v.z; d[3] = (_Float16)v.w;
  }
  __syncthreads();

  // ---- phase 1: logits = QL @ Kchunk^T -> f16 fragments in sLg -----------
  for (int ch = 0; ch < 8; ++ch) {
    if (ch < 7) __builtin_prefetch((const void*)(Kb4 + (size_t)(ch + 1) * 1024 + t * 4), 0, 1);
#pragma unroll
    for (int n = 0; n < 4; ++n) {                           // stage K chunk, scaled, row-major
      const int i4 = t + 256 * n, row = i4 >> 4, c4 = (i4 & 15) * 4;
      const float m = mask[(size_t)b * kS + sp * kSP + ch * 64 + row] * kScale;
      const float4 v = Kb4[(size_t)ch * 1024 + i4];
      _Float16* d = sKV + row * kLDH + c4;
      d[0] = (_Float16)(v.x * m); d[1] = (_Float16)(v.y * m);
      d[2] = (_Float16)(v.z * m); d[3] = (_Float16)(v.w * m);
    }
    __syncthreads();
#pragma unroll
    for (int i = 0; i < 2; ++i) {
      const int tm = tmB + i;
      v8f acc = {0.f, 0.f, 0.f, 0.f, 0.f, 0.f, 0.f, 0.f};
#pragma unroll
      for (int k0 = 0; k0 < 64; k0 += 32) {
        v16h a  = ldA_f16(sQL + tm * 16 * kLDH + k0, kLDH, lane);
        v16h bb = ldBT_f16(sKV + tn * 16 * kLDH + k0, kLDH, lane);
        acc = WMMA_F16(a, bb, acc);
      }
      stC_f16(sLg + tm * 16 * kLDL + ch * 64 + tn * 16, kLDL, lane, acc);
    }
    __syncthreads();
  }

  // ---- phase 2: row max / exp / sumexp over 512 cols (4 threads/row) -----
  const int row = t >> 2, part = t & 3;
  {
    float mx = -1e30f;
    for (int j = part * 128; j < part * 128 + 128; ++j)
      mx = fmaxf(mx, (float)sLg[row * kLDL + j]);
    sRed[row * 4 + part] = mx;
  }
  __syncthreads();
  const float rowm = fmaxf(fmaxf(sRed[row * 4 + 0], sRed[row * 4 + 1]),
                           fmaxf(sRed[row * 4 + 2], sRed[row * 4 + 3]));
  __syncthreads();
  {
    float ss = 0.f;
    for (int j = part * 128; j < part * 128 + 128; ++j) {
      const float e = __expf((float)sLg[row * kLDL + j] - rowm);
      sLg[row * kLDL + j] = (_Float16)e;
      ss += e;
    }
    sRed[row * 4 + part] = ss;
  }
  __syncthreads();
  if (part == 0) {
    sM[row] = rowm;
    sL[row] = sRed[row * 4 + 0] + sRed[row * 4 + 1] + sRed[row * 4 + 2] + sRed[row * 4 + 3];
  }
  __syncthreads();

  // ---- phase 3: Acc = P @ Vsplit (K = 512, V staged transposed) ----------
  v8f acc0 = {0.f, 0.f, 0.f, 0.f, 0.f, 0.f, 0.f, 0.f};
  v8f acc1 = {0.f, 0.f, 0.f, 0.f, 0.f, 0.f, 0.f, 0.f};
  for (int ch = 0; ch < 8; ++ch) {
    if (ch < 7) __builtin_prefetch((const void*)(Vb4 + (size_t)(ch + 1) * 1024 + t * 4), 0, 1);
    __syncthreads();
#pragma unroll
    for (int n = 0; n < 4; ++n) {                           // stage V chunk transposed
      const int i4 = t + 256 * n, r2 = i4 >> 4, c4 = (i4 & 15) * 4;
      const float4 v = Vb4[(size_t)ch * 1024 + i4];
      sKV[(c4 + 0) * kLDH + r2] = (_Float16)v.x;
      sKV[(c4 + 1) * kLDH + r2] = (_Float16)v.y;
      sKV[(c4 + 2) * kLDH + r2] = (_Float16)v.z;
      sKV[(c4 + 3) * kLDH + r2] = (_Float16)v.w;
    }
    __syncthreads();
#pragma unroll
    for (int k0 = 0; k0 < 64; k0 += 32) {
      v16h a0 = ldA_f16(sLg + (tmB + 0) * 16 * kLDL + ch * 64 + k0, kLDL, lane);
      v16h a1 = ldA_f16(sLg + (tmB + 1) * 16 * kLDL + ch * 64 + k0, kLDL, lane);
      v16h bb = ldBT_f16(sKV + tn * 16 * kLDH + k0, kLDH, lane);
      acc0 = WMMA_F16(a0, bb, acc0);
      acc1 = WMMA_F16(a1, bb, acc1);
    }
  }
  if (t < 64) {
    MS[((size_t)bh * kNS + sp) * 64 + t] = sM[t];
    LS[((size_t)bh * kNS + sp) * 64 + t] = sL[t];
  }
  __syncthreads();                                          // all reads of sLg done
  stC_f32(sStg + (tmB + 0) * 16 * kLDF + tn * 16, kLDF, lane, acc0);
  stC_f32(sStg + (tmB + 1) * 16 * kLDF + tn * 16, kLDF, lane, acc1);
  __syncthreads();
  {
    float4* dst4 = (float4*)(CP + ((size_t)bh * kNS + sp) * 4096);
#pragma unroll
    for (int n = 0; n < 4; ++n) {
      const int i4 = t + 256 * n, r2 = i4 >> 4, c4 = (i4 & 15) * 4;
      const float* s = sStg + r2 * kLDF + c4;
      dst4[i4] = make_float4(s[0], s[1], s[2], s[3]);
    }
  }
}

// ---------------------------------------------------------------------------
// Kernel 3b: LSE-merge the 8 split partials -> c [64x64] per bh (float4 IO).
// ---------------------------------------------------------------------------
__global__ __launch_bounds__(256) void k_merge(const float* __restrict__ CP,
                                               const float* __restrict__ MS,
                                               const float* __restrict__ LS,
                                               float* __restrict__ C) {
  __shared__ float sm[kNS][64];
  __shared__ float sl[kNS][64];
  __shared__ float sMx[64];
  __shared__ float sDen[64];
  const int bh = blockIdx.x, t = threadIdx.x;
  for (int idx = t; idx < kNS * 64; idx += 256) {
    const int i = idx >> 6, r = idx & 63;
    sm[i][r] = MS[((size_t)bh * kNS + i) * 64 + r];
    sl[i][r] = LS[((size_t)bh * kNS + i) * 64 + r];
  }
  __syncthreads();
  if (t < 64) {
    float M = sm[0][t];
#pragma unroll
    for (int i = 1; i < kNS; ++i) M = fmaxf(M, sm[i][t]);
    float den = 0.f;
#pragma unroll
    for (int i = 0; i < kNS; ++i) den += __expf(sm[i][t] - M) * sl[i][t];
    sMx[t] = M;
    sDen[t] = den;
  }
  __syncthreads();
  float4* C4 = (float4*)(C + (size_t)bh * 4096);
  for (int i4 = t; i4 < 1024; i4 += 256) {
    const int r = i4 >> 4;
    float sx = 0.f, sy = 0.f, sz = 0.f, sw = 0.f;
#pragma unroll
    for (int i = 0; i < kNS; ++i) {
      const float w = __expf(sm[i][r] - sMx[r]);
      const float4 cp = ((const float4*)(CP + ((size_t)bh * kNS + i) * 4096))[i4];
      sx += w * cp.x; sy += w * cp.y; sz += w * cp.z; sw += w * cp.w;
    }
    const float inv = 1.0f / sDen[r];
    C4[i4] = make_float4(sx * inv, sy * inv, sz * inv, sw * inv);
  }
}

// ---------------------------------------------------------------------------
// Kernel 4 (dominant pass): per 64-row Q tile:
//   P = softmax_rows(Q@KL^T) ; B1 = P@invK2 ; X = B1@c.
// invK2 / c staged transposed; P and B1 kept as f16 fragments.
// ---------------------------------------------------------------------------
__global__ __launch_bounds__(256) void k_main(const float* __restrict__ Q,
                                              const float* __restrict__ mask,
                                              const float* __restrict__ KL,
                                              const float* __restrict__ INV,
                                              const float* __restrict__ Cm,
                                              float* __restrict__ out) {
  extern __shared__ char smem[];
  float*    sP   = (float*)smem;                     // 64 x kLDF f32 (logits / X stage)
  _Float16* sPh  = (_Float16*)(sP + 64 * kLDF);      // 64 x 72 (P f16)
  _Float16* sB1h = sPh + 64 * kLDH;                  // 64 x 72 (B1 f16)
  _Float16* sQ   = sB1h + 64 * kLDH;                 // 64 x 72 (Q row-major)
  _Float16* sKL  = sQ + 64 * kLDH;                   // 64 x 72 (KL row-major)
  _Float16* sIvT = sKL + 64 * kLDH;                  // 64 x 72 (invK2 transposed)
  _Float16* sCT  = sIvT + 64 * kLDH;                 // 64 x 72 (c transposed)
  const int bh = blockIdx.x >> 6, tile = blockIdx.x & 63, b = bh / kH;
  const int t = threadIdx.x, lane = t & 31, wave = t >> 5;
  const int tn = wave & 3, tmB = (wave >> 2) * 2;

  const float4* Q4  = (const float4*)(Q + ((size_t)bh * kS + (size_t)tile * 64) * kD);
  const float4* KL4 = (const float4*)(KL + (size_t)bh * 4096);
  const float4* IV4 = (const float4*)(INV + (size_t)bh * 4096);
  const float4* Cm4 = (const float4*)(Cm + (size_t)bh * 4096);

#pragma unroll
  for (int n = 0; n < 4; ++n) {
    const int i4 = t + 256 * n, row = i4 >> 4, c4 = (i4 & 15) * 4;
    const float m = mask[(size_t)b * kS + tile * 64 + row] * kScale;
    const float4 q = Q4[i4];
    _Float16* dq = sQ + row * kLDH + c4;
    dq[0] = (_Float16)(q.x * m); dq[1] = (_Float16)(q.y * m);
    dq[2] = (_Float16)(q.z * m); dq[3] = (_Float16)(q.w * m);
    const float4 kl = KL4[i4];
    _Float16* dk = sKL + row * kLDH + c4;
    dk[0] = (_Float16)kl.x; dk[1] = (_Float16)kl.y; dk[2] = (_Float16)kl.z; dk[3] = (_Float16)kl.w;
    const float4 iv = IV4[i4];
    sIvT[(c4 + 0) * kLDH + row] = (_Float16)iv.x;
    sIvT[(c4 + 1) * kLDH + row] = (_Float16)iv.y;
    sIvT[(c4 + 2) * kLDH + row] = (_Float16)iv.z;
    sIvT[(c4 + 3) * kLDH + row] = (_Float16)iv.w;
    const float4 cc = Cm4[i4];
    sCT[(c4 + 0) * kLDH + row] = (_Float16)cc.x;
    sCT[(c4 + 1) * kLDH + row] = (_Float16)cc.y;
    sCT[(c4 + 2) * kLDH + row] = (_Float16)cc.z;
    sCT[(c4 + 3) * kLDH + row] = (_Float16)cc.w;
  }
  __syncthreads();

  // P = Qtile @ KL^T  (f32 logits for softmax)
#pragma unroll
  for (int i = 0; i < 2; ++i) {
    const int tm = tmB + i;
    v8f acc = {0.f, 0.f, 0.f, 0.f, 0.f, 0.f, 0.f, 0.f};
#pragma unroll
    for (int k0 = 0; k0 < 64; k0 += 32) {
      v16h a  = ldA_f16(sQ + tm * 16 * kLDH + k0, kLDH, lane);
      v16h bb = ldBT_f16(sKL + tn * 16 * kLDH + k0, kLDH, lane);
      acc = WMMA_F16(a, bb, acc);
    }
    stC_f32(sP + tm * 16 * kLDF + tn * 16, kLDF, lane, acc);
  }
  __syncthreads();
  if (t < 64) {                                   // row softmax -> f16 P
    float mx = -1e30f;
    for (int j = 0; j < 64; ++j) mx = fmaxf(mx, sP[t * kLDF + j]);
    float sm = 0.f;
    float e[64];
    for (int j = 0; j < 64; ++j) { e[j] = __expf(sP[t * kLDF + j] - mx); sm += e[j]; }
    const float inv = 1.0f / sm;
    for (int j = 0; j < 64; ++j) sPh[t * kLDH + j] = (_Float16)(e[j] * inv);
  }
  __syncthreads();

  // B1 = P @ invK2  (B from transposed stage -> contiguous)
#pragma unroll
  for (int i = 0; i < 2; ++i) {
    const int tm = tmB + i;
    v8f acc = {0.f, 0.f, 0.f, 0.f, 0.f, 0.f, 0.f, 0.f};
#pragma unroll
    for (int k0 = 0; k0 < 64; k0 += 32) {
      v16h a  = ldA_f16(sPh + tm * 16 * kLDH + k0, kLDH, lane);
      v16h bb = ldBT_f16(sIvT + tn * 16 * kLDH + k0, kLDH, lane);
      acc = WMMA_F16(a, bb, acc);
    }
    stC_f16(sB1h + tm * 16 * kLDH + tn * 16, kLDH, lane, acc);
  }
  __syncthreads();

  // X = B1 @ c  -> f32 stage -> float4 global store
#pragma unroll
  for (int i = 0; i < 2; ++i) {
    const int tm = tmB + i;
    v8f acc = {0.f, 0.f, 0.f, 0.f, 0.f, 0.f, 0.f, 0.f};
#pragma unroll
    for (int k0 = 0; k0 < 64; k0 += 32) {
      v16h a  = ldA_f16(sB1h + tm * 16 * kLDH + k0, kLDH, lane);
      v16h bb = ldBT_f16(sCT + tn * 16 * kLDH + k0, kLDH, lane);
      acc = WMMA_F16(a, bb, acc);
    }
    stC_f32(sP + tm * 16 * kLDF + tn * 16, kLDF, lane, acc);
  }
  __syncthreads();
  {
    float4* out4 = (float4*)(out + ((size_t)bh * kS + (size_t)tile * 64) * kD);
#pragma unroll
    for (int n = 0; n < 4; ++n) {
      const int i4 = t + 256 * n, row = i4 >> 4, c4 = (i4 & 15) * 4;
      const float* s = sP + row * kLDF + c4;
      out4[i4] = make_float4(s[0], s[1], s[2], s[3]);
    }
  }
}

// ---------------------------------------------------------------------------
extern "C" void kernel_launch(void* const* d_in, const int* in_sizes, int n_in,
                              void* d_out, int out_size, void* d_ws, size_t ws_size,
                              hipStream_t stream) {
  const float* Q    = (const float*)d_in[0];
  const float* K    = (const float*)d_in[1];
  const float* V    = (const float*)d_in[2];
  const float* mask = (const float*)d_in[3];
  float* out = (float*)d_out;
  float* ws  = (float*)d_ws;
  (void)in_sizes; (void)n_in; (void)out_size; (void)ws_size;   // ws need: ~13.8 MB

  // reset global colsum-max scalar (graph-capture safe)
  hipMemsetAsync((char*)d_ws + WS_GM * sizeof(float), 0, sizeof(float), stream);

  k_landmarks<<<kBH * kL, 64, 0, stream>>>(Q, K, mask, ws + WS_QL, ws + WS_KL);

  const size_t sh2a = (3 * 64 * 66 + 64) * sizeof(float);
  k_kernel2<<<kBH, 256, sh2a, stream>>>(ws + WS_QL, ws + WS_KL, ws + WS_A2,
                                        (int*)(ws + WS_GM));

  const size_t sh2b = (size_t)5 * 64 * 66 * sizeof(float);
  k_inverse<<<kBH, 256, sh2b, stream>>>(ws + WS_A2, ws + WS_GM, ws + WS_INV);

  const size_t sh3 = (size_t)64 * kLDL * sizeof(_Float16) + (256 + 128) * sizeof(float) +
                     (size_t)2 * 64 * kLDH * sizeof(_Float16);
  k_kernel3<<<dim3(kBH, kNS), 256, sh3, stream>>>(K, V, mask, ws + WS_QL,
                                                  ws + WS_CP, ws + WS_MS, ws + WS_LS);

  k_merge<<<kBH, 256, 0, stream>>>(ws + WS_CP, ws + WS_MS, ws + WS_LS, ws + WS_C);

  const size_t sh4 = (size_t)64 * kLDF * sizeof(float) +
                     (size_t)6 * 64 * kLDH * sizeof(_Float16);
  k_main<<<kBH * (kS / 64), 256, sh4, stream>>>(Q, mask, ws + WS_KL, ws + WS_INV,
                                                ws + WS_C, out);
}
